// QuantizedLinear_61933428408793
// MI455X (gfx1250) — matrix operand order repaired and checked by reading, not compiled, not run on hardware
//
#include <hip/hip_runtime.h>

// ---------------------------------------------------------------------------
// QuantizedLinear: y[m,n] = sum_k x[m,k] * sign(W[n,k]) + bias[n]
// M = 16*4096 = 65536, K = 512, N = 512, f32 in/out.
// Strategy: f16 WMMA (sign(W) exact in f16), f32 accumulate.
//   - x tile (128x512) converted f32->f16 into LDS once (x read once from HBM)
//   - sign(W) pre-packed into B-fragment order; staged per 64-col chunk into
//     LDS with double-buffered ASYNC global->LDS copies (CDNA5 ASYNCcnt path)
// ---------------------------------------------------------------------------

typedef _Float16 v16h __attribute__((ext_vector_type(16)));
typedef _Float16 v8h  __attribute__((ext_vector_type(8)));
typedef float    v8f  __attribute__((ext_vector_type(8)));
typedef int      v4i  __attribute__((ext_vector_type(4)));

typedef __attribute__((address_space(1))) v4i* gv4i_p;   // global int4*
typedef __attribute__((address_space(3))) v4i* lv4i_p;   // LDS int4*

static constexpr int IN_F  = 512;
static constexpr int OUT_F = 512;
static constexpr int MTOT  = 16 * 4096;          // 65536 rows
static constexpr int KT    = IN_F / 32;          // 16 K-slabs of 32
static constexpr int NSUB  = OUT_F / 16;         // 32 column subtiles of 16
static constexpr int M_TILE = 128;               // rows per block
static constexpr int RS     = IN_F + 8;          // padded LDS row stride (halves)
static constexpr int CHUNKH = 4 * KT * 32 * 16;  // halves per 64-col B chunk (32768)

#if __has_builtin(__builtin_amdgcn_global_load_async_to_lds_b128)
#define QL_HAVE_ASYNC 1
#else
#define QL_HAVE_ASYNC 0
#endif

__device__ __forceinline__ void ql_wait_async0() {
#if __has_builtin(__builtin_amdgcn_s_wait_asynccnt)
    __builtin_amdgcn_s_wait_asynccnt(0);
#else
    asm volatile("s_wait_asynccnt 0x0" ::: "memory");
#endif
}

__device__ __forceinline__ void ql_async_b128(const _Float16* g, _Float16* l) {
#if QL_HAVE_ASYNC
    __builtin_amdgcn_global_load_async_to_lds_b128(
        (gv4i_p)(unsigned long long)g,
        (lv4i_p)(unsigned)(unsigned long long)l,
        0, 0);
#else
    *(uint4*)l = *(const uint4*)g;
#endif
}

// Packed B layout (halves): [nsub=32][kt=16][lane=32][h=16]
//   lane L holds column nsub*16 + (L%16),
//   halves h=0..15 are K = kt*32 + (L/16)*16 + h.
__global__ __launch_bounds__(256)
void QL_pack_weight(const float* __restrict__ w, _Float16* __restrict__ bp) {
    int idx  = blockIdx.x * blockDim.x + threadIdx.x;   // 0..16383
    int lane = idx & 31;
    int kt   = (idx >> 5) & 15;
    int nsub = idx >> 9;
    int n     = nsub * 16 + (lane & 15);
    int kbase = kt * 32 + ((lane >> 4) << 4);
    const float* wr = w + (long)n * IN_F + kbase;
    _Float16* outp  = bp + (long)idx * 16;
#pragma unroll
    for (int h = 0; h < 16; ++h) {
        float v = wr[h];
        float s = (v > 0.0f) ? 1.0f : ((v < 0.0f) ? -1.0f : 0.0f);
        outp[h] = (_Float16)s;
    }
}

__global__ __launch_bounds__(256)
void QL_gemm(const float* __restrict__ x, const _Float16* __restrict__ bp,
             const float* __restrict__ bias, float* __restrict__ y) {
    __shared__ __attribute__((aligned(64))) _Float16 sA[M_TILE * RS];   // 130 KB
    __shared__ __attribute__((aligned(64))) _Float16 sB[2 * CHUNKH];    // 128 KB (double-buffered)

    const int  t    = threadIdx.x;
    const int  wave = t >> 5;
    const int  lane = t & 31;
    const long rowBlock = (long)blockIdx.x * M_TILE;

#if QL_HAVE_ASYNC
    // Kick off async staging of B chunk 0 into buffer 0 (overlaps x-tile load)
    {
        const _Float16* bg = bp;
        _Float16* dst = sB;
#pragma unroll
        for (int j = 0; j < 16; ++j) {
            int o = (j * 256 + t) * 8;           // halves (16 bytes)
            ql_async_b128(bg + o, dst + o);
        }
    }
#endif

    // ---- Prologue: load 128x512 f32 x-tile, convert to f16 in LDS ----
    const float* xg = x + rowBlock * IN_F;
#pragma unroll 2
    for (int it = 0; it < 128; ++it) {
        int flat = it * 1024 + t * 4;            // global-linear within tile
        float4 v = *(const float4*)(xg + flat);
        int row = flat >> 9;                     // /512
        int col = flat & 511;
        union { _Float16 h[4]; unsigned long long u; } pk;
        pk.h[0] = (_Float16)v.x; pk.h[1] = (_Float16)v.y;
        pk.h[2] = (_Float16)v.z; pk.h[3] = (_Float16)v.w;
        *(unsigned long long*)(sA + row * RS + col) = pk.u;
    }
#if QL_HAVE_ASYNC
    ql_wait_async0();
#endif
    __syncthreads();

    const int arow = wave * 16 + (lane & 15);
    const int koff = (lane >> 4) << 4;           // 0 or 16 halves

    for (int nc = 0; nc < 8; ++nc) {             // 8 N-chunks of 64 columns
        const int cur = nc & 1;

#if QL_HAVE_ASYNC
        // Prefetch next chunk into the other buffer (its last readers finished
        // at iteration nc-1, before that iteration's wait+barrier).
        if (nc + 1 < 8) {
            const _Float16* bg = bp + (long)(nc + 1) * CHUNKH;
            _Float16* dst = sB + (1 - cur) * CHUNKH;
#pragma unroll
            for (int j = 0; j < 16; ++j) {
                int o = (j * 256 + t) * 8;
                ql_async_b128(bg + o, dst + o);
            }
        }
#else
        // Synchronous staging fallback: coalesced 16B copies of chunk nc.
        {
            const _Float16* bg = bp + (long)nc * CHUNKH;
            _Float16* dst = sB + cur * CHUNKH;
#pragma unroll
            for (int j = 0; j < 16; ++j) {
                int o = (j * 256 + t) * 8;
                *(uint4*)(dst + o) = *(const uint4*)(bg + o);
            }
            __syncthreads();
        }
#endif

        const _Float16* sBc = sB + cur * CHUNKH;
        v8f acc[4] = {v8f{}, v8f{}, v8f{}, v8f{}};

#pragma unroll
        for (int kt = 0; kt < KT; ++kt) {
            // A fragment: lanes 0-15 -> K {0..15}; lanes 16-31 -> K {16..31}
            union { v16h v; v8h p[2]; } a;
            const _Float16* ap = sA + arow * RS + kt * 32 + koff;
            a.p[0] = *(const v8h*)ap;
            a.p[1] = *(const v8h*)(ap + 8);

            const _Float16* bbase = sBc + ((long)kt * 32 + lane) * 16;
            v16h b0 = *(const v16h*)(bbase + 0 * (KT * 32 * 16));
            v16h b1 = *(const v16h*)(bbase + 1 * (KT * 32 * 16));
            v16h b2 = *(const v16h*)(bbase + 2 * (KT * 32 * 16));
            v16h b3 = *(const v16h*)(bbase + 3 * (KT * 32 * 16));

            acc[0] = __builtin_amdgcn_wmma_f32_16x16x32_f16(false, a.v, false, b0,
                                                            (short)0, acc[0], false, false);
            acc[1] = __builtin_amdgcn_wmma_f32_16x16x32_f16(false, a.v, false, b1,
                                                            (short)0, acc[1], false, false);
            acc[2] = __builtin_amdgcn_wmma_f32_16x16x32_f16(false, a.v, false, b2,
                                                            (short)0, acc[2], false, false);
            acc[3] = __builtin_amdgcn_wmma_f32_16x16x32_f16(false, a.v, false, b3,
                                                            (short)0, acc[3], false, false);
        }

        // ---- Epilogue: C/D layout VGPR r -> M=(lane/16)*8+r, N=lane%16 ----
        const long r0 = rowBlock + wave * 16 + ((lane >> 4) << 3);
#pragma unroll
        for (int j = 0; j < 4; ++j) {
            int col  = nc * 64 + j * 16 + (lane & 15);
            float bv = bias[col];
            float* yp = y + r0 * OUT_F + col;
#pragma unroll
            for (int r = 0; r < 8; ++r)
                yp[(long)r * OUT_F] = acc[j][r] + bv;
        }

#if QL_HAVE_ASYNC
        ql_wait_async0();      // next chunk's data fully landed in LDS
#endif
        __syncthreads();       // make it visible / protect buffers
    }
}

extern "C" void kernel_launch(void* const* d_in, const int* in_sizes, int n_in,
                              void* d_out, int out_size, void* d_ws, size_t ws_size,
                              hipStream_t stream) {
    const float* x    = (const float*)d_in[0];   // [16,4096,512] f32
    const float* w    = (const float*)d_in[1];   // [512,512]     f32
    const float* bias = (const float*)d_in[2];   // [512]         f32
    float* y = (float*)d_out;                    // [16,4096,512] f32

    _Float16* bp = (_Float16*)d_ws;              // 512 KB packed sign(W) in f16

    QL_pack_weight<<<(NSUB * KT * 32) / 256, 256, 0, stream>>>(w, bp);
    QL_gemm<<<MTOT / M_TILE, 256, 0, stream>>>(x, bp, bias, y);
}
